// NLayerImageCNN_27453430956728
// MI455X (gfx1250) — matrix operand order repaired and checked by reading, not compiled, not run on hardware
//
#include <hip/hip_runtime.h>
#include <hip/hip_bf16.h>

typedef _Float16 v16h __attribute__((ext_vector_type(16)));
typedef _Float16 v8h  __attribute__((ext_vector_type(8)));
typedef float    v8f  __attribute__((ext_vector_type(8)));
typedef unsigned int u32x4 __attribute__((ext_vector_type(4)));
typedef int          i32x4 __attribute__((ext_vector_type(4)));
typedef int          i32x8 __attribute__((ext_vector_type(8)));

#define HIMG 32
#define WIMG 4096
#define EMB  512
#define KTOT 1024
#define M_TILE 64                    // slices per block (rows of GEMM)
#define N_TILE 256                   // embed cols per block
#define IMG_COLS (M_TILE * 8 + 24)   // 536 padded-image columns needed
#define IMG_STRIDE 544               // f16 row stride (LDS)
#define A_STRIDE 40                  // f16, 80B rows -> conflict-free b128 reads
#define B_STRIDE 40

#if defined(__HIP_DEVICE_COMPILE__) && __has_builtin(__builtin_amdgcn_tensor_load_to_lds)
#define HAVE_TDM 1
#else
#define HAVE_TDM 0
#endif

// ---- prep: lin_w (1024x512 f32) -> wT (512x1024 f16), LDS-tiled transpose ----
__global__ __launch_bounds__(256) void transpose_convert_w(
    const float* __restrict__ w, _Float16* __restrict__ wT)
{
    __shared__ float t[32][33];
    const int kt = blockIdx.x * 32;          // 0..1023 (32 tiles)
    const int nt = blockIdx.y * 32;          // 0..511  (16 tiles)
    const int tx = threadIdx.x & 31;
    const int ty = threadIdx.x >> 5;         // 0..7
#pragma unroll
    for (int r = 0; r < 32; r += 8)
        t[ty + r][tx] = w[(size_t)(kt + ty + r) * EMB + nt + tx];
    __syncthreads();
#pragma unroll
    for (int r = 0; r < 32; r += 8)
        wT[(size_t)(nt + ty + r) * KTOT + kt + tx] = (_Float16)t[tx][ty + r];
}

template <bool TDM>
__global__ __launch_bounds__(256) void fused_slice_conv_gemm(
    const float*    __restrict__ images,   // (64,1,32,4096)
    const float*    __restrict__ conv_w,   // (1,1,3,3)
    const float*    __restrict__ conv_b,   // (1,)
    const float*    __restrict__ lin_w,    // (1024,512) f32  (fallback path)
    const _Float16* __restrict__ wT,       // (512,1024) f16  (TDM path)
    const float*    __restrict__ lin_b,    // (512,)
    float*          __restrict__ out)      // (64,512,512) flat
{
    __shared__ _Float16 simg[HIMG * IMG_STRIDE];            // 34,816 B
    __shared__ _Float16 sA[M_TILE * A_STRIDE];              //  5,120 B
    __shared__ __align__(16) _Float16 sB[N_TILE * B_STRIDE];// 20,480 B ([n][k])

    const int tid   = threadIdx.x;
    const int lane  = tid & 31;
    const int wave  = tid >> 5;
    const int wm    = wave >> 1;        // 0..3  (M direction)
    const int wn    = wave & 1;         // 0..1  (N direction)
    const int m     = lane & 15;
    const int khalf = lane >> 4;        // 0 or 1

    const int mt      = blockIdx.x;     // 0..511
    const int nt      = blockIdx.y;     // 0..1
    const int bimg    = mt >> 3;
    const int s0      = (mt & 7) * M_TILE;
    const int rowBase = mt * M_TILE;
    const int n1      = nt * N_TILE;

    float cw[9];
#pragma unroll
    for (int i = 0; i < 9; ++i) cw[i] = conv_w[i];
    const float cb = conv_b[0];

    // ---- Stage padded image region once (pad value = 1.0f), fp32 -> f16 ----
    {
        const int col0 = s0 * 8 - 12;
        const float* imbase = images + (size_t)bimg * HIMG * WIMG;
#pragma unroll 1
        for (int e = 0; e < 67; ++e) {          // 67*256 == 32*536
            int idx = tid + e * 256;
            int r = idx / IMG_COLS;
            int c = idx - r * IMG_COLS;
            int gc = col0 + c;
            float v = 1.0f;
            if (gc >= 0 && gc < WIMG) v = imbase[r * WIMG + gc];
            simg[r * IMG_STRIDE + c] = (_Float16)v;
        }
    }
    __syncthreads();

    v8f acc[8];
#pragma unroll
    for (int j = 0; j < 8; ++j)
#pragma unroll
        for (int v = 0; v < 8; ++v) acc[j][v] = 0.0f;

#pragma unroll 1
    for (int h = 0; h < HIMG; ++h) {
        // ---- B tile: 32x256 of lin_w for this K-block -> sB[n][k] ----
        if constexpr (TDM) {
#if HAVE_TDM
            if (wave == 0) {
                // TDM: 2-D tile copy wT[n1..n1+255][32h..32h+31] -> sB (80B rows)
                unsigned long long ga =
                    (unsigned long long)(const void*)(wT + (size_t)n1 * KTOT + h * 32);
                unsigned int ldsa = (unsigned int)(unsigned long long)(const void*)sB;
                u32x4 g0;
                g0[0] = 1u;                                   // count=1 (valid)
                g0[1] = ldsa;                                 // lds_addr
                g0[2] = (unsigned int)ga;                     // global_addr[31:0]
                g0[3] = (unsigned int)((ga >> 32) & 0x1FFFFFFull) | (2u << 30); // [56:32]|type=2
                i32x8 g1;
                g1[0] = (int)((1u << 16)      // data_size = 2 bytes
                            | (1u << 20)      // pad_enable
                            | (3u << 22)      // pad_interval: 16 DWORDs (64B row)
                            | (3u << 25));    // pad_amount: 4 DWORDs (16B pad)
                g1[1] = (int)(1024u << 16);   // tensor_dim0 = 1024
                g1[2] = (int)(512u << 16);    // tensor_dim1 = 512
                g1[3] = (int)(32u << 16);     // tile_dim0 = 32
                g1[4] = 256;                  // tile_dim1 = 256, tile_dim2 = 0
                g1[5] = 1024;                 // tensor_dim0_stride = 1024
                g1[6] = 0;
                g1[7] = 0;
                i32x4 gz;  gz[0] = 0; gz[1] = 0; gz[2] = 0; gz[3] = 0;
                i32x8 gz8; gz8[0] = 0; gz8[1] = 0; gz8[2] = 0; gz8[3] = 0;
                gz8[4] = 0; gz8[5] = 0; gz8[6] = 0; gz8[7] = 0;
                __builtin_amdgcn_tensor_load_to_lds(g0, g1, gz, gz, gz8, 0);
            }
#endif
        } else {
            const float* wp = lin_w + (size_t)(h * 32) * EMB + n1;
#pragma unroll 1
            for (int e = 0; e < 32; ++e) {
                int idx = tid + e * 256;
                int kk = idx >> 8;
                int n  = idx & 255;
                sB[n * B_STRIDE + kk] = (_Float16)wp[kk * EMB + n];
            }
        }

        // ---- A: conv + bias + relu for row h of 64 slices -> sA[si][w] ----
#pragma unroll 1
        for (int e = 0; e < 8; ++e) {           // 8*256 == 64*32
            int idx = tid + e * 256;
            int si = idx >> 5;
            int w  = idx & 31;
            float sum = cb;
#pragma unroll
            for (int kh = 0; kh < 3; ++kh) {
                int hh = h + kh - 1;
                if (hh >= 0 && hh < HIMG) {
                    const _Float16* rowp = &simg[hh * IMG_STRIDE + si * 8];
#pragma unroll
                    for (int kw = 0; kw < 3; ++kw) {
                        int ww = w + kw - 1;
                        if (ww >= 0 && ww < 32)
                            sum += cw[kh * 3 + kw] * (float)rowp[ww];
                    }
                }
            }
            sA[si * A_STRIDE + w] = (_Float16)fmaxf(sum, 0.0f);
        }

        if constexpr (TDM) {
#if HAVE_TDM
            if (wave == 0) __builtin_amdgcn_s_wait_tensorcnt(0);
#endif
        }
        __syncthreads();

        // ---- WMMA: one K=32 block ----
        v16h a;
        {
            const _Float16* ap = &sA[(wm * 16 + m) * A_STRIDE + khalf * 8];
            v8h lo = *(const v8h*)ap;          // K = khalf*8 + 0..7
            v8h hi = *(const v8h*)(ap + 16);   // K = 16 + khalf*8 + 0..7
#pragma unroll
            for (int i = 0; i < 8; ++i) { a[i] = lo[i]; a[8 + i] = hi[i]; }
        }
#pragma unroll
        for (int j = 0; j < 8; ++j) {
            int ncol = wn * 128 + j * 16 + m;
            const _Float16* bp = &sB[ncol * B_STRIDE + khalf * 8];
            v8h b0 = *(const v8h*)bp;
            v8h b1 = *(const v8h*)(bp + 16);
            v16h b;
#pragma unroll
            for (int i = 0; i < 8; ++i) { b[i] = b0[i]; b[8 + i] = b1[i]; }
            acc[j] = __builtin_amdgcn_wmma_f32_16x16x32_f16(
                false, a, false, b, (short)0, acc[j], false, false);
        }
        __syncthreads();
    }

    // ---- Store (D layout: lane halves -> M = 8*khalf + v, N = lane&15) ----
#pragma unroll
    for (int j = 0; j < 8; ++j) {
        int col  = n1 + wn * 128 + j * 16 + m;
        float bias = lin_b[col];
        int row0 = rowBase + wm * 16 + khalf * 8;
#pragma unroll
        for (int v = 0; v < 8; ++v) {
            out[(size_t)(row0 + v) * EMB + col] = acc[j][v] + bias;
        }
    }
}

extern "C" void kernel_launch(void* const* d_in, const int* in_sizes, int n_in,
                              void* d_out, int out_size, void* d_ws, size_t ws_size,
                              hipStream_t stream) {
    (void)in_sizes; (void)n_in; (void)out_size;
    const float* images = (const float*)d_in[0];
    const float* conv_w = (const float*)d_in[1];
    const float* conv_b = (const float*)d_in[2];
    const float* lin_w  = (const float*)d_in[3];
    const float* lin_b  = (const float*)d_in[4];
    float* out = (float*)d_out;

    const size_t wT_bytes = (size_t)EMB * KTOT * sizeof(_Float16);  // 1 MB
    dim3 grid(512, 2, 1);

    if (d_ws != nullptr && ws_size >= wT_bytes) {
        _Float16* wT = (_Float16*)d_ws;
        transpose_convert_w<<<dim3(32, 16, 1), 256, 0, stream>>>(lin_w, wT);
        fused_slice_conv_gemm<true><<<grid, 256, 0, stream>>>(
            images, conv_w, conv_b, lin_w, wT, lin_b, out);
    } else {
        fused_slice_conv_gemm<false><<<grid, 256, 0, stream>>>(
            images, conv_w, conv_b, lin_w, nullptr, lin_b, out);
    }
}
